// GATBN_17222818857484
// MI455X (gfx1250) — matrix operand order repaired and checked by reading, not compiled, not run on hardware
//
#include <hip/hip_runtime.h>
#include <hip/hip_fp16.h>
#include <stdint.h>

#define NN 100000
#define EE 1600000
#define INC 128
#define HCD 64     // H*C
#define EPSV 1e-5f

typedef __attribute__((ext_vector_type(16))) _Float16 v16h;
typedef __attribute__((ext_vector_type(8)))  float    v8f;
typedef __attribute__((ext_vector_type(8)))  unsigned v8u;
typedef __attribute__((ext_vector_type(4)))  unsigned v4u;
typedef __attribute__((ext_vector_type(4)))  int      v4i;

#if defined(__has_builtin)
#  if __has_builtin(__builtin_amdgcn_global_load_async_to_lds_b128)
#    define HAVE_ASYNC_LDS 1
#  endif
#  if __has_builtin(__builtin_amdgcn_s_wait_asynccnt)
#    define HAVE_WAIT_ASYNC 1
#  endif
#endif

// 16-byte global->LDS copy; async path when the toolchain exposes it.
static __device__ __forceinline__ void cp16_async(const unsigned* g, unsigned* l) {
#if defined(HAVE_ASYNC_LDS)
    __builtin_amdgcn_global_load_async_to_lds_b128(
        (__attribute__((address_space(1))) v4i*)(unsigned long long)(uintptr_t)g,
        (__attribute__((address_space(3))) v4i*)(unsigned)(uintptr_t)l,
        0, 0);
#else
    *(v4u*)l = *(const v4u*)g;
#endif
}
static __device__ __forceinline__ void async_join() {
#if defined(HAVE_ASYNC_LDS)
#  if defined(HAVE_WAIT_ASYNC)
    __builtin_amdgcn_s_wait_asynccnt(0);
#  else
    asm volatile("s_wait_asynccnt 0x0" ::: "memory");
#  endif
#endif
}

static __device__ __forceinline__ unsigned pk2h(float a, float b) {
    _Float16 ha = (_Float16)a, hb = (_Float16)b;
    unsigned short ua = __builtin_bit_cast(unsigned short, ha);
    unsigned short ub = __builtin_bit_cast(unsigned short, hb);
    return (unsigned)ua | ((unsigned)ub << 16);
}

// order-preserving float<->uint for atomicMax-based segment max
static __device__ __forceinline__ unsigned fflip(float f) {
    unsigned u = __builtin_bit_cast(unsigned, f);
    return u ^ (unsigned)(((int)u >> 31) | 0x80000000);
}
static __device__ __forceinline__ float funflip(unsigned u) {
    unsigned m = (u & 0x80000000u) ? 0x80000000u : 0xFFFFFFFFu;
    return __builtin_bit_cast(float, u ^ m);
}

static __device__ __forceinline__ v8f wmma_f16(v16h a, v16h b, v8f c) {
    return __builtin_amdgcn_wmma_f32_16x16x32_f16(false, a, false, b, (short)0, c, false, false);
}

// ---------------- K0: init scratch ----------------
__global__ void k_init(unsigned* lmax, float* denom, float* nstats, float* estats) {
    int t = blockIdx.x * blockDim.x + threadIdx.x;
    int tot = NN * 4;
    if (t < tot) { lmax[t] = 0u; denom[t] = 0.f; }
    if (t < 128) nstats[t] = 0.f;
    if (t < 64)  estats[t] = 0.f;
}

// ---------------- K1a: pack weights (f16 K-pair packed) ----------------
__global__ void k_pack_w(const float* Wl, const float* Wr, const float* Wres,
                         const float* We, const float* Weo,
                         unsigned* Wn_pk, unsigned* We_pk, unsigned* Weo_pk) {
    int t = threadIdx.x;
    for (int idx = t; idx < 64 * 192; idx += 256) {        // [K/2=64][192]
        int k2 = idx / 192, n = idx % 192;
        const float* W = (n < 64) ? Wl : (n < 128 ? Wr : Wres);
        int c = n & 63;
        Wn_pk[idx] = pk2h(W[(2 * k2) * 64 + c], W[(2 * k2 + 1) * 64 + c]);
    }
    for (int idx = t; idx < 16 * 64; idx += 256) {         // [K/2=16][64]
        int k2 = idx >> 6, c = idx & 63;
        We_pk[idx] = pk2h(We[(2 * k2) * 64 + c], We[(2 * k2 + 1) * 64 + c]);
    }
    for (int idx = t; idx < 32 * 32; idx += 256) {         // [K/2=32][32]
        int k2 = idx >> 5, c = idx & 31;
        Weo_pk[idx] = pk2h(Weo[(2 * k2) * 32 + c], Weo[(2 * k2 + 1) * 32 + c]);
    }
}

// ---------------- K1b: pack x to f16 pairs ----------------
__global__ void k_pack_x(const float* x, unsigned* xpk) {
    long i = (long)blockIdx.x * blockDim.x + threadIdx.x;
    if (i >= (long)NN * 64) return;
    long row = i >> 6; int k2 = (int)(i & 63);
    xpk[i] = pk2h(x[row * 128 + 2 * k2], x[row * 128 + 2 * k2 + 1]);
}

// ---------------- K2: node GEMM  x @ [W_l|W_r|W_res] ----------------
__global__ __launch_bounds__(256) void k_node_gemm(const unsigned* xpk, const unsigned* Wn_pk,
                                                   float* xl, float* xr, float* accp) {
    __shared__ __align__(16) unsigned ldsX[32 * 65];      // 32 rows, 64 k2-dwords, pad 65
    __shared__ __align__(16) unsigned ldsW[64 * 196];     // 64 k2-rows, 192 cols, pad 196 (784B rows: 16B aligned)
    int t = threadIdx.x;
    int wave = t >> 5, lane = t & 31, half = lane >> 4, l15 = lane & 15;
    long rowBase = (long)blockIdx.x * 32;

    for (int i = t; i < 64 * 48; i += 256) {              // 3072 x 16B async chunks
        int r = i / 48, c4 = (i % 48) * 4;
        cp16_async(&Wn_pk[r * 192 + c4], &ldsW[r * 196 + c4]);
    }
    for (int i = t; i < 32 * 64; i += 256) {
        int r = i >> 6, c = i & 63;
        ldsX[r * 65 + c] = xpk[(rowBase + r) * 64 + c];
    }
    async_join();
    __syncthreads();

    int mtile = wave & 1;              // 2 row-tiles of 16
    int nt0 = (wave >> 1) * 3;         // 3 col-tiles of 16 each (12 total = 192 cols)
    v8f ac0 = {}, ac1 = {}, ac2 = {};
    for (int ks = 0; ks < 4; ++ks) {   // K=128 -> 4 steps of 32
        v8u a;
        int abase = (mtile * 16 + l15) * 65 + ks * 16 + half * 4;
        #pragma unroll
        for (int i = 0; i < 8; ++i) a[i] = ldsX[abase + (i & 3) + ((i >> 2) * 8)];
        v16h av = __builtin_bit_cast(v16h, a);
        #pragma unroll
        for (int j = 0; j < 3; ++j) {
            int col = (nt0 + j) * 16 + l15;
            v8u b;
            #pragma unroll
            for (int v = 0; v < 8; ++v) b[v] = ldsW[(ks * 16 + half * 4 + (v & 3) + ((v >> 2) * 8)) * 196 + col];
            v16h bv = __builtin_bit_cast(v16h, b);
            if (j == 0) ac0 = wmma_f16(av, bv, ac0);
            else if (j == 1) ac1 = wmma_f16(av, bv, ac1);
            else ac2 = wmma_f16(av, bv, ac2);
        }
    }
    long gRow0 = rowBase + mtile * 16 + half * 8;
    #pragma unroll
    for (int j = 0; j < 3; ++j) {
        int colg = (nt0 + j) * 16 + l15;
        #pragma unroll
        for (int r = 0; r < 8; ++r) {
            float vv = (j == 0) ? ac0[r] : (j == 1) ? ac1[r] : ac2[r];
            long row = gRow0 + r;
            if (colg < 64)       xl[row * 64 + colg] = vv;
            else if (colg < 128) xr[row * 64 + (colg - 64)] = vv;
            else                 accp[row * 64 + (colg - 128)] = vv;   // residual init
        }
    }
}

// ---------------- K3: edge GEMM -> m, logits, segment max ----------------
__global__ __launch_bounds__(256) void k_edge_m(const float* eattr, const unsigned* We_pk,
                                                const int* ei, const float* xl, const float* xr,
                                                const float* att, _Float16* mh, float* logits,
                                                unsigned* lmax) {
    __shared__ __align__(16) unsigned ldsE[128 * 17];   // 128 edges x 16 k2-dwords (K=32)
    __shared__ __align__(16) unsigned ldsW[16 * 68];    // 272B rows: 16B aligned
    __shared__ __align__(16) int ldsSrc[128];
    __shared__ __align__(16) int ldsDst[128];
    __shared__ float ldsAtt[64];
    int t = threadIdx.x, wave = t >> 5, lane = t & 31, half = lane >> 4, l15 = lane & 15;
    long eBase = (long)blockIdx.x * 128;

    for (int i = t; i < 16 * 16; i += 256) {            // weight tile: async 16B chunks
        int r = i >> 4, c4 = (i & 15) * 4;
        cp16_async(&We_pk[r * 64 + c4], &ldsW[r * 68 + c4]);
    }
    if (t < 32)      cp16_async((const unsigned*)&ei[eBase + (long)t * 4], (unsigned*)&ldsSrc[t * 4]);
    else if (t < 64) cp16_async((const unsigned*)&ei[(long)EE + eBase + (long)(t - 32) * 4],
                                (unsigned*)&ldsDst[(t - 32) * 4]);
    for (int i = t; i < 128 * 16; i += 256) {
        int r = i >> 4, c2 = i & 15;
        const float* p = eattr + (eBase + r) * 32 + c2 * 2;
        ldsE[r * 17 + c2] = pk2h(p[0], p[1]);
    }
    if (t < 64) ldsAtt[t] = att[t];
    async_join();
    __syncthreads();

    int erow = wave * 16;  // wave handles 16 edges x all 64 cols (4 n-tiles = 4 heads)
    v8u a;
    int abase = (erow + l15) * 17 + half * 4;
    #pragma unroll
    for (int i = 0; i < 8; ++i) a[i] = ldsE[abase + (i & 3) + ((i >> 2) * 8)];
    v16h av = __builtin_bit_cast(v16h, a);

    v8f acc4[4] = {v8f{}, v8f{}, v8f{}, v8f{}};
    #pragma unroll
    for (int nt = 0; nt < 4; ++nt) {
        v8u b;
        #pragma unroll
        for (int v = 0; v < 8; ++v) b[v] = ldsW[(half * 4 + (v & 3) + ((v >> 2) * 8)) * 68 + nt * 16 + l15];
        v16h bv = __builtin_bit_cast(v16h, b);
        acc4[nt] = wmma_f16(av, bv, acc4[nt]);
    }

    #pragma unroll
    for (int nt = 0; nt < 4; ++nt) {          // nt == head (C==16)
        float attv = ldsAtt[nt * 16 + l15];
        #pragma unroll
        for (int r = 0; r < 8; ++r) {
            int le = erow + half * 8 + r;
            long ge = eBase + le;
            int col = nt * 16 + l15;
            int s = ldsSrc[le], d = ldsDst[le];
            float mv = acc4[nt][r] + xl[(long)s * 64 + col] + xr[(long)d * 64 + col];
            mh[ge * 64 + col] = (_Float16)mv;
            float p = fmaxf(mv, 0.f) * attv;  // relu (neg_slope==0) dot att
            p += __shfl_xor(p, 1, 16);
            p += __shfl_xor(p, 2, 16);
            p += __shfl_xor(p, 4, 16);
            p += __shfl_xor(p, 8, 16);
            if (l15 == 0) {
                logits[ge * 4 + nt] = p;
                atomicMax(&lmax[(long)d * 4 + nt], fflip(p));
            }
        }
    }
}

// ---------------- K4: softmax denominator ----------------
__global__ void k_denom(const float* logits, const unsigned* lmax, const int* ei, float* denom) {
    long t = (long)blockIdx.x * blockDim.x + threadIdx.x;
    if (t >= (long)EE * 4) return;
    long e = t >> 2; int hh = (int)(t & 3);
    int d = ei[(long)EE + e];
    float ex = __expf(logits[t] - funflip(lmax[(long)d * 4 + hh]));
    atomicAdd(&denom[(long)d * 4 + hh], ex);
}

// ---------------- K5: scatter alpha * xl[src] into acc ----------------
__global__ __launch_bounds__(256) void k_scatter(const float* logits, const unsigned* lmax,
                                                 const float* denom, const int* ei,
                                                 const float* xl, float* accp) {
    int t = threadIdx.x, wave = t >> 5, lane = t & 31;
    long eBase = (long)blockIdx.x * 64 + wave * 8;
    int c0 = lane, c1 = lane + 32;
    int h0 = lane >> 4, h1 = h0 + 2;
    for (int i = 0; i < 8; ++i) {
        long e = eBase + i;
        int s = ei[e], d = ei[(long)EE + e];
        float a0 = __expf(logits[e * 4 + h0] - funflip(lmax[(long)d * 4 + h0])) / denom[(long)d * 4 + h0];
        float a1 = __expf(logits[e * 4 + h1] - funflip(lmax[(long)d * 4 + h1])) / denom[(long)d * 4 + h1];
        atomicAdd(&accp[(long)d * 64 + c0], a0 * xl[(long)s * 64 + c0]);
        atomicAdd(&accp[(long)d * 64 + c1], a1 * xl[(long)s * 64 + c1]);
    }
}

// ---------------- K6: e_out GEMM (m @ W_eo) + edge-norm stats ----------------
__global__ __launch_bounds__(256) void k_eout(const unsigned* mh32, const unsigned* Weo_pk,
                                              float* eo, float* estats) {
    __shared__ __align__(16) unsigned ldsM[128 * 33];   // 128 edges x 32 k2-dwords (K=64)
    __shared__ __align__(16) unsigned ldsW[32 * 36];    // 144B rows: 16B aligned
    __shared__ float ldsS[64];
    int t = threadIdx.x, wave = t >> 5, lane = t & 31, half = lane >> 4, l15 = lane & 15;
    long eBase = (long)blockIdx.x * 128;
    if (t < 64) ldsS[t] = 0.f;
    for (int i = t; i < 32 * 8; i += 256) {             // weight tile: async 16B chunks
        int r = i >> 3, c4 = (i & 7) * 4;
        cp16_async(&Weo_pk[r * 32 + c4], &ldsW[r * 36 + c4]);
    }
    for (int i = t; i < 128 * 32; i += 256) ldsM[(i >> 5) * 33 + (i & 31)] = mh32[eBase * 32 + i];
    async_join();
    __syncthreads();

    v8f acc2[2] = {v8f{}, v8f{}};
    for (int ks = 0; ks < 2; ++ks) {
        v8u a;
        int abase = (wave * 16 + l15) * 33 + ks * 16 + half * 4;
        #pragma unroll
        for (int i = 0; i < 8; ++i) a[i] = ldsM[abase + (i & 3) + ((i >> 2) * 8)];
        v16h av = __builtin_bit_cast(v16h, a);
        #pragma unroll
        for (int nt = 0; nt < 2; ++nt) {
            v8u b;
            #pragma unroll
            for (int v = 0; v < 8; ++v) b[v] = ldsW[(ks * 16 + half * 4 + (v & 3) + ((v >> 2) * 8)) * 36 + nt * 16 + l15];
            v16h bv = __builtin_bit_cast(v16h, b);
            acc2[nt] = wmma_f16(av, bv, acc2[nt]);
        }
    }
    #pragma unroll
    for (int nt = 0; nt < 2; ++nt) {
        int col = nt * 16 + l15;
        float s1 = 0.f, s2 = 0.f;
        #pragma unroll
        for (int r = 0; r < 8; ++r) {
            float vv = acc2[nt][r];
            long ge = eBase + wave * 16 + half * 8 + r;
            eo[ge * 32 + col] = vv;
            s1 += vv; s2 += vv * vv;
        }
        atomicAdd(&ldsS[col], s1);
        atomicAdd(&ldsS[32 + col], s2);
    }
    __syncthreads();
    if (t < 64) atomicAdd(&estats[t], ldsS[t]);
}

// ---------------- K7: node-norm stats ----------------
__global__ void k_nstats(const float* accp, float* nstats) {
    __shared__ float lds[128];
    int t = threadIdx.x;
    if (t < 128) lds[t] = 0.f;
    __syncthreads();
    long tot = (long)NN * 64;
    for (long i = (long)blockIdx.x * 256 + t; i < tot; i += (long)gridDim.x * 256) {
        float v = accp[i];
        int c = (int)(i & 63);
        atomicAdd(&lds[c], v);
        atomicAdd(&lds[64 + c], v * v);
    }
    __syncthreads();
    if (t < 128) atomicAdd(&nstats[t], lds[t]);
}

// ---------------- K8: node GraphNorm + relu -> d_out ----------------
__global__ void k_nnorm(const float* accp, const float* nstats, const float* w,
                        const float* b, const float* ms, float* xo) {
    long i = (long)blockIdx.x * blockDim.x + threadIdx.x;
    if (i >= (long)NN * 64) return;
    int c = (int)(i & 63);
    float mu = nstats[c] * (1.f / NN);
    float ex2 = nstats[64 + c] * (1.f / NN);
    float msv = ms[c];
    float var = ex2 - 2.f * msv * mu * mu + msv * msv * mu * mu;
    float sub = accp[i] - msv * mu;
    xo[i] = fmaxf(w[c] * sub * rsqrtf(var + EPSV) + b[c], 0.f);
}

// ---------------- K9: edge GraphNorm + relu (in-place on d_out) ----------------
__global__ void k_enorm(float* eo, const float* estats, const float* w,
                        const float* b, const float* ms) {
    long i = (long)blockIdx.x * blockDim.x + threadIdx.x;
    if (i >= (long)EE * 32) return;
    int c = (int)(i & 31);
    float mu = estats[c] * (1.f / EE);
    float ex2 = estats[32 + c] * (1.f / EE);
    float msv = ms[c];
    float var = ex2 - 2.f * msv * mu * mu + msv * msv * mu * mu;
    float sub = eo[i] - msv * mu;
    eo[i] = fmaxf(w[c] * sub * rsqrtf(var + EPSV) + b[c], 0.f);
}

extern "C" void kernel_launch(void* const* d_in, const int* in_sizes, int n_in,
                              void* d_out, int out_size, void* d_ws, size_t ws_size,
                              hipStream_t stream) {
    const float* x     = (const float*)d_in[0];
    const int*   ei    = (const int*)d_in[1];
    const float* eattr = (const float*)d_in[2];
    const float* Wl    = (const float*)d_in[3];
    const float* Wr    = (const float*)d_in[4];
    const float* We    = (const float*)d_in[5];
    const float* att   = (const float*)d_in[6];
    const float* Wres  = (const float*)d_in[7];
    const float* Weo   = (const float*)d_in[8];
    const float* nbw   = (const float*)d_in[9];
    const float* nbb   = (const float*)d_in[10];
    const float* nbms  = (const float*)d_in[11];
    const float* ebw   = (const float*)d_in[12];
    const float* ebb   = (const float*)d_in[13];
    const float* ebms  = (const float*)d_in[14];

    char* ws = (char*)d_ws;
    size_t off = 0;
    auto alloc = [&](size_t bytes) -> void* {
        void* p = ws + off;
        off = (off + bytes + 255) & ~(size_t)255;
        return p;
    };
    unsigned* xpk    = (unsigned*)alloc((size_t)NN * 64 * 4);
    unsigned* Wn_pk  = (unsigned*)alloc(64 * 192 * 4);
    unsigned* We_pk  = (unsigned*)alloc(16 * 64 * 4);
    unsigned* Weo_pk = (unsigned*)alloc(32 * 32 * 4);
    float*    xl     = (float*)alloc((size_t)NN * 64 * 4);
    float*    xr     = (float*)alloc((size_t)NN * 64 * 4);
    float*    accp   = (float*)alloc((size_t)NN * 64 * 4);
    _Float16* mh     = (_Float16*)alloc((size_t)EE * 64 * 2);
    float*    logits = (float*)alloc((size_t)EE * 4 * 4);
    unsigned* lmax   = (unsigned*)alloc((size_t)NN * 4 * 4);
    float*    denom  = (float*)alloc((size_t)NN * 4 * 4);
    float*    nstats = (float*)alloc(128 * 4);
    float*    estats = (float*)alloc(64 * 4);

    float* xo = (float*)d_out;             // [NN*64]
    float* eo = xo + (size_t)NN * 64;      // [EE*32]

    k_init<<<(NN * 4 + 255) / 256, 256, 0, stream>>>(lmax, denom, nstats, estats);
    k_pack_w<<<1, 256, 0, stream>>>(Wl, Wr, Wres, We, Weo, Wn_pk, We_pk, Weo_pk);
    k_pack_x<<<(NN * 64 + 255) / 256, 256, 0, stream>>>(x, xpk);
    k_node_gemm<<<NN / 32, 256, 0, stream>>>(xpk, Wn_pk, xl, xr, accp);
    k_edge_m<<<EE / 128, 256, 0, stream>>>(eattr, We_pk, ei, xl, xr, att, mh, logits, lmax);
    k_denom<<<(EE * 4) / 256, 256, 0, stream>>>(logits, lmax, ei, denom);
    k_scatter<<<EE / 64, 256, 0, stream>>>(logits, lmax, denom, ei, xl, accp);
    k_eout<<<EE / 128, 256, 0, stream>>>((const unsigned*)mh, Weo_pk, eo, estats);
    k_nstats<<<1024, 256, 0, stream>>>(accp, nstats);
    k_nnorm<<<(NN * 64) / 256, 256, 0, stream>>>(accp, nstats, nbw, nbb, nbms, xo);
    k_enorm<<<(EE * 32) / 256, 256, 0, stream>>>(eo, estats, ebw, ebb, ebms);
}
